// MultiHeadAttention_27178553049486
// MI455X (gfx1250) — matrix operand order repaired and checked by reading, not compiled, not run on hardware
//
#include <hip/hip_runtime.h>
#include <hip/hip_bf16.h>

typedef __attribute__((ext_vector_type(16))) __bf16 v16bf;
typedef __attribute__((ext_vector_type(8)))  __bf16 v8bf;
typedef __attribute__((ext_vector_type(8)))  float  v8f;

#define WMMA_BF16(a, b, c) \
    __builtin_amdgcn_wmma_f32_16x16x32_bf16(false, (a), false, (b), (short)0, (c), false, false)

__device__ inline v16bf make_frag(const __bf16* lo, const __bf16* hi) {
    v8bf a = *(const v8bf*)lo;
    v8bf b = *(const v8bf*)hi;
    return __builtin_shufflevector(a, b, 0, 1, 2, 3, 4, 5, 6, 7, 8, 9, 10, 11, 12, 13, 14, 15);
}

// async memory->LDS copy of 16B (+16B at offset), CDNA5 ASYNCcnt path
__device__ inline void async_copy_b128(const void* gsrc, void* ldsDst) {
    unsigned lds = (unsigned)(unsigned long long)(uintptr_t)ldsDst;
    asm volatile("global_load_async_to_lds_b128 %0, %1, off"
                 :: "v"(lds), "v"((unsigned long long)(uintptr_t)gsrc) : "memory");
}
__device__ inline void async_copy_b128x2(const void* gsrc, void* ldsDst) {
    unsigned lds = (unsigned)(unsigned long long)(uintptr_t)ldsDst;
    unsigned long long g = (unsigned long long)(uintptr_t)gsrc;
    asm volatile("global_load_async_to_lds_b128 %0, %1, off" :: "v"(lds), "v"(g) : "memory");
    asm volatile("global_load_async_to_lds_b128 %0, %1, off offset:16" :: "v"(lds), "v"(g) : "memory");
}
__device__ inline void wait_async() {
    asm volatile("s_wait_asynccnt 0" ::: "memory");
}

// ---------------------------------------------------------------------------
// fp32 -> bf16 conversion
// ---------------------------------------------------------------------------
__global__ void f32_to_bf16_kernel(const float* __restrict__ src, __bf16* __restrict__ dst, int n) {
    int i = blockIdx.x * blockDim.x + threadIdx.x;
    if (i < n) dst[i] = (__bf16)src[i];
}

// decoder_output fp32 [8192,1024] -> right half of concat bf16 [8192,2048]
__global__ void dec_to_concat_kernel(const float* __restrict__ dec, __bf16* __restrict__ cat) {
    int i = blockIdx.x * blockDim.x + threadIdx.x;   // 0 .. 8388608
    int row = i >> 10;
    int col = i & 1023;
    cat[(size_t)row * 2048 + 1024 + col] = (__bf16)dec[i];
}

// ---------------------------------------------------------------------------
// C = A @ W^T + bias ; A[Mtot,Kdim] bf16 row-major, W[Ntot,Kdim] bf16 row-major
// TANH=0: store bf16; TANH=1: store tanh(x) as f32
// block: 256 threads (8 waves), tile 64(M) x 128(N), K-step 32
// Tiles staged via GLOBAL_LOAD_ASYNC_TO_LDS_B128 (ASYNCcnt).
// ---------------------------------------------------------------------------
template <int TANH>
__global__ void __launch_bounds__(256)
wmma_gemm_bias_kernel(const __bf16* __restrict__ A, const __bf16* __restrict__ W,
                      const float* __restrict__ bias, void* outPtr,
                      int Mtot, int Ntot, int Kdim) {
    __shared__ __align__(16) __bf16 As[64][32];
    __shared__ __align__(16) __bf16 Ws[128][32];

    const int tid     = threadIdx.x;
    const int lane    = tid & 31;
    const int wid     = tid >> 5;       // 0..7
    const int waveM   = wid >> 2;       // 0..1
    const int waveN   = wid & 3;        // 0..3
    const int lr      = lane & 15;
    const int halfSel = lane >> 4;      // 0 or 1
    const int mBase   = blockIdx.y * 64;
    const int nBase   = blockIdx.x * 128;

    // per-thread staging coordinates (invariant over K loop)
    const int aRow = tid >> 2;          // 0..63
    const int aCol = (tid & 3) * 8;     // 0,8,16,24
    const int wRow = tid >> 1;          // 0..127
    const int wCol = (tid & 1) * 16;    // 0,16

    v8f acc[2][2] = {};

    for (int k0 = 0; k0 < Kdim; k0 += 32) {
        // stage A tile 64x32 and W tile 128x32 with async LDS DMA
        async_copy_b128(A + (size_t)(mBase + aRow) * Kdim + k0 + aCol, &As[aRow][aCol]);
        async_copy_b128x2(W + (size_t)(nBase + wRow) * Kdim + k0 + wCol, &Ws[wRow][wCol]);
        wait_async();
        __syncthreads();

        v16bf afrag[2], bfrag[2];
#pragma unroll
        for (int i = 0; i < 2; i++) {
            int row = waveM * 32 + i * 16 + lr;
            int kb  = halfSel * 8;                       // A-layout: runs [kb,kb+8) and [kb+16,kb+24)
            afrag[i] = make_frag(&As[row][kb], &As[row][kb + 16]);
        }
#pragma unroll
        for (int j = 0; j < 2; j++) {
            int row = waveN * 32 + j * 16 + lr;
            int kb2 = halfSel * 8;                       // B-layout: runs [kb2,kb2+8) and [kb2+16,kb2+24)
            bfrag[j] = make_frag(&Ws[row][kb2], &Ws[row][kb2 + 16]);
        }
#pragma unroll
        for (int i = 0; i < 2; i++)
#pragma unroll
            for (int j = 0; j < 2; j++)
                acc[i][j] = WMMA_BF16(afrag[i], bfrag[j], acc[i][j]);
        __syncthreads();
    }

    // epilogue: D layout row = r + 8*halfSel, col = lr  (branch-free, templated)
#pragma unroll
    for (int i = 0; i < 2; i++) {
#pragma unroll
        for (int j = 0; j < 2; j++) {
            int   colG = nBase + waveN * 32 + j * 16 + lr;
            float bv   = bias[colG];
#pragma unroll
            for (int r = 0; r < 8; r++) {
                int   rowG = mBase + waveM * 32 + i * 16 + r + halfSel * 8;
                float v    = acc[i][j][r] + bv;
                if constexpr (TANH == 1)
                    ((float*)outPtr)[(size_t)rowG * Ntot + colG] = tanhf(v);
                else
                    ((__bf16*)outPtr)[(size_t)rowG * Ntot + colG] = (__bf16)v;
            }
        }
    }
}

// ---------------------------------------------------------------------------
// Flash attention: one block = 4 waves = 64 q rows for one (b, h).
// Qb [8192,1024] bf16, Kb [16384,1024] bf16 (V == K). Context written into
// left half of concat bf16 [8192,2048].
// ---------------------------------------------------------------------------
__global__ void __launch_bounds__(128)
flash_attn_kernel(const __bf16* __restrict__ Qb, const __bf16* __restrict__ Kb,
                  __bf16* __restrict__ cat) {
    __shared__ __align__(16) __bf16 vt[128][32];   // V^T tile: [d][key]
    __shared__ __align__(16) __bf16 pT[4][16][32]; // per-wave P tile (row-major)

    const int tid     = threadIdx.x;   // 0..127
    const int lane    = tid & 31;
    const int wid     = tid >> 5;      // 0..3
    const int lr      = lane & 15;
    const int halfSel = lane >> 4;

    const int b     = blockIdx.z;
    const int h     = blockIdx.y;
    const int qBase = blockIdx.x * 64 + wid * 16;

    // preload 16x128 Q tile as 4 A-fragments (kept in VGPRs all loop)
    v16bf qf[4];
    {
        const __bf16* qrow = Qb + ((size_t)(b * 1024 + qBase + lr)) * 1024 + h * 128;
#pragma unroll
        for (int f = 0; f < 4; f++) {
            int kb = f * 32 + halfSel * 8;
            qf[f]  = make_frag(qrow + kb, qrow + kb + 16);
        }
    }

    float mrow[8], lrow[8];
    v8f   o[8] = {};
#pragma unroll
    for (int r = 0; r < 8; r++) { mrow[r] = -3.0e38f; lrow[r] = 0.0f; }

    for (int s = 0; s < 64; s++) {
        const int keyBase = s * 32;

        // prefetch next 32-key K/V tile while this one computes
        if (s + 1 < 64) {
            const __bf16* pf = Kb + ((size_t)(b * 2048 + keyBase + 32 + (tid >> 2))) * 1024
                                  + h * 128 + (tid & 3) * 32;
            __builtin_prefetch(pf, 0, 1);
        }

        __syncthreads();   // previous iteration done reading vt
        // cooperative transposed stage of V tile [32 keys x 128 d] -> vt[d][key]
#pragma unroll
        for (int it = 0; it < 4; it++) {
            int idx = tid + it * 128;
            int key = idx >> 4;         // 0..31
            int dc  = idx & 15;         // chunk of 8 d's
            const __bf16* src = Kb + ((size_t)(b * 2048 + keyBase + key)) * 1024 + h * 128 + dc * 8;
            v8bf v = *(const v8bf*)src;
#pragma unroll
            for (int i = 0; i < 8; i++) vt[dc * 8 + i][key] = v[i];
        }
        __syncthreads();

        // S = Q @ K^T for 32 keys: two 16-key N tiles, kdim 128 = 4 steps
        v8f s0 = {}, s1 = {};
        {
            const size_t kRow0 = (size_t)(b * 2048 + keyBase + lr) * 1024;
            const size_t kRow1 = kRow0 + (size_t)16 * 1024;
#pragma unroll
            for (int f = 0; f < 4; f++) {
                int   off = h * 128 + f * 32 + halfSel * 8;
                v16bf b0  = make_frag(Kb + kRow0 + off, Kb + kRow0 + off + 16);
                v16bf b1  = make_frag(Kb + kRow1 + off, Kb + kRow1 + off + 16);
                s0 = WMMA_BF16(qf[f], b0, s0);
                s1 = WMMA_BF16(qf[f], b1, s1);
            }
        }

        // online softmax: row reductions across the 16-lane column group
        float scale[8];
#pragma unroll
        for (int r = 0; r < 8; r++) {
            float t = fmaxf(s0[r], s1[r]);
            t = fmaxf(t, __shfl_xor(t, 1, 32));
            t = fmaxf(t, __shfl_xor(t, 2, 32));
            t = fmaxf(t, __shfl_xor(t, 4, 32));
            t = fmaxf(t, __shfl_xor(t, 8, 32));
            float mn = fmaxf(mrow[r], t);
            scale[r] = __expf(mrow[r] - mn);
            mrow[r]  = mn;
        }
#pragma unroll
        for (int r = 0; r < 8; r++) {
            float p0 = __expf(s0[r] - mrow[r]);
            float p1 = __expf(s1[r] - mrow[r]);
            pT[wid][r + halfSel * 8][lr]      = (__bf16)p0;   // D->row-major in LDS
            pT[wid][r + halfSel * 8][16 + lr] = (__bf16)p1;
            float rs = p0 + p1;
            rs += __shfl_xor(rs, 1, 32);
            rs += __shfl_xor(rs, 2, 32);
            rs += __shfl_xor(rs, 4, 32);
            rs += __shfl_xor(rs, 8, 32);
            lrow[r] = lrow[r] * scale[r] + rs;
        }
#pragma unroll
        for (int t = 0; t < 8; t++)
#pragma unroll
            for (int r = 0; r < 8; r++) o[t][r] *= scale[r];

        asm volatile("s_wait_dscnt 0" ::: "memory");  // P stores -> P fragment loads (same wave)

        const int kb = halfSel * 8;
        v16bf pfr = make_frag(&pT[wid][lr][kb], &pT[wid][lr][kb + 16]);

        // O += P @ V : 8 d-tiles of 16, kdim 32 (one WMMA each)
#pragma unroll
        for (int t = 0; t < 8; t++) {
            int   d0  = t * 16 + lr;
            int   kb2 = halfSel * 8;
            v16bf vf  = make_frag(&vt[d0][kb2], &vt[d0][kb2 + 16]);
            o[t] = WMMA_BF16(pfr, vf, o[t]);
        }
    }

    // normalize and write context into left half of concat
    float inv[8];
#pragma unroll
    for (int r = 0; r < 8; r++) inv[r] = 1.0f / lrow[r];
#pragma unroll
    for (int t = 0; t < 8; t++) {
#pragma unroll
        for (int r = 0; r < 8; r++) {
            int rowG = b * 1024 + qBase + r + halfSel * 8;
            int colG = h * 128 + t * 16 + lr;
            cat[(size_t)rowG * 2048 + colG] = (__bf16)(o[t][r] * inv[r]);
        }
    }
}

// ---------------------------------------------------------------------------
extern "C" void kernel_launch(void* const* d_in, const int* in_sizes, int n_in,
                              void* d_out, int out_size, void* d_ws, size_t ws_size,
                              hipStream_t stream) {
    (void)in_sizes; (void)n_in; (void)out_size; (void)ws_size;

    const float* dec = (const float*)d_in[0];  // [8,1024,1024]
    const float* enc = (const float*)d_in[1];  // [8,2048,1024]
    const float* Wq  = (const float*)d_in[2];  // [1024,1024]
    const float* bq  = (const float*)d_in[3];
    const float* Wk  = (const float*)d_in[4];
    const float* bk  = (const float*)d_in[5];
    const float* Wo  = (const float*)d_in[6];  // [1024,2048]
    const float* bo  = (const float*)d_in[7];

    char* ws = (char*)d_ws;
    __bf16* decb = (__bf16*)(ws);                    // 8192*1024   (16 MiB)
    __bf16* encb = (__bf16*)(ws + 16777216ull);      // 16384*1024  (32 MiB)
    __bf16* wqb  = (__bf16*)(ws + 50331648ull);      // 1024*1024   ( 2 MiB)
    __bf16* wkb  = (__bf16*)(ws + 52428800ull);      // 1024*1024   ( 2 MiB)
    __bf16* wob  = (__bf16*)(ws + 54525952ull);      // 1024*2048   ( 4 MiB)
    __bf16* qb   = (__bf16*)(ws + 58720256ull);      // 8192*1024   (16 MiB)
    __bf16* kb   = (__bf16*)(ws + 75497472ull);      // 16384*1024  (32 MiB)
    __bf16* cat  = (__bf16*)(ws + 109051904ull);     // 8192*2048   (32 MiB)

    // 1) precision-convert everything to bf16
    f32_to_bf16_kernel<<<8388608 / 256, 256, 0, stream>>>(dec, decb, 8388608);
    f32_to_bf16_kernel<<<16777216 / 256, 256, 0, stream>>>(enc, encb, 16777216);
    f32_to_bf16_kernel<<<1048576 / 256, 256, 0, stream>>>(Wq, wqb, 1048576);
    f32_to_bf16_kernel<<<1048576 / 256, 256, 0, stream>>>(Wk, wkb, 1048576);
    f32_to_bf16_kernel<<<2097152 / 256, 256, 0, stream>>>(Wo, wob, 2097152);
    dec_to_concat_kernel<<<8388608 / 256, 256, 0, stream>>>(dec, cat);

    // 2) Q = dec @ Wq^T + bq   (M=8192, N=1024, K=1024) -> bf16
    wmma_gemm_bias_kernel<0><<<dim3(1024 / 128, 8192 / 64), 256, 0, stream>>>(
        decb, wqb, bq, (void*)qb, 8192, 1024, 1024);
    // 3) K = enc @ Wk^T + bk   (M=16384, N=1024, K=1024) -> bf16
    wmma_gemm_bias_kernel<0><<<dim3(1024 / 128, 16384 / 64), 256, 0, stream>>>(
        encb, wkb, bk, (void*)kb, 16384, 1024, 1024);

    // 4) attention (V = K), context -> left half of concat
    flash_attn_kernel<<<dim3(16, 8, 8), 128, 0, stream>>>(qb, kb, cat);

    // 5) out = tanh(concat @ Wo^T + bo)  (M=8192, N=1024, K=2048) -> f32 d_out
    wmma_gemm_bias_kernel<1><<<dim3(1024 / 128, 8192 / 64), 256, 0, stream>>>(
        cat, wob, bo, d_out, 8192, 1024, 2048);
}
